// StandardGNNModel_4380866642060
// MI455X (gfx1250) — matrix operand order repaired and checked by reading, not compiled, not run on hardware
//
#include <hip/hip_runtime.h>
#include <hip/hip_bf16.h>

#define N_NODES 100000

typedef __attribute__((ext_vector_type(16))) _Float16 v16h;
typedef __attribute__((ext_vector_type(8)))  _Float16 v8h;
typedef __attribute__((ext_vector_type(4)))  _Float16 v4h;
typedef __attribute__((ext_vector_type(8)))  float    v8f;

// ---------------------------------------------------------------- utilities

__global__ void gcn_zero_f32(float* __restrict__ p, int n) {
  int i = blockIdx.x * blockDim.x + threadIdx.x;
  if (i < n) p[i] = 0.0f;
}

// deg[dst]++ for every edge (self loops folded in later as +1)
__global__ void gcn_deg_count(const long long* __restrict__ dst, long long E,
                              float* __restrict__ deg) {
  long long i = (long long)blockIdx.x * blockDim.x + threadIdx.x;
  if (i < E) atomicAdd(&deg[(int)dst[i]], 1.0f);
}

// dis[i] = rsqrt(deg[i] + 1)   (+1 = self loop; always > 0)
__global__ void gcn_deg_rsqrt(float* __restrict__ deg, int n) {
  int i = blockIdx.x * blockDim.x + threadIdx.x;
  if (i < n) deg[i] = rsqrtf(deg[i] + 1.0f);
}

// f32 -> f16 elementwise, 4-wide
__global__ void gcn_cvt_f16(const float* __restrict__ src,
                            _Float16* __restrict__ dst, long long n4) {
  long long i = (long long)blockIdx.x * blockDim.x + threadIdx.x;
  if (i >= n4) return;
  float4 v = ((const float4*)src)[i];
  v4h o;
  o.x = (_Float16)v.x; o.y = (_Float16)v.y;
  o.z = (_Float16)v.z; o.w = (_Float16)v.w;
  ((v4h*)dst)[i] = o;
}

// Repack W (K x N row-major f32) into this kernel's contiguous-k B-fragment order (f16).
// Fragment linear index: (((ntile*KT + ktile)*32 + lane)*16 + j)
//   lane<16 : k = ktile*32 + j,      n = ntile*16 + lane
//   lane>=16: k = ktile*32 + 16 + j, n = ntile*16 + (lane-16)
__global__ void gcn_repack_w(const float* __restrict__ W,
                             _Float16* __restrict__ Bp, int K, int N) {
  int idx = blockIdx.x * blockDim.x + threadIdx.x;
  if (idx >= K * N) return;
  int KT   = K >> 5;
  int j    = idx & 15;
  int lane = (idx >> 4) & 31;
  int rest = idx >> 9;
  int kt   = rest % KT;
  int nt   = rest / KT;
  int k = kt * 32 + (lane >> 4) * 16 + j;
  int n = nt * 16 + (lane & 15);
  Bp[idx] = (_Float16)W[(size_t)k * N + n];
}

// ------------------------------------------------------------------- GEMM
// Hg(MxN,f32) = A(MxK,f16 row-major) * Bpacked;  Ha = Hg * dis[row]^2
// (self-loop term fused into the epilogue).  One wave -> 16x64 strip:
// 4 x v_wmma_f32_16x16x32_f16 accumulators, A fragment reused 4x, and the
// K loop is compile-time unrolled with explicit double buffering so next
// iteration's loads overlap the current WMMAs.
template <int KT>
__global__ void gcn_gemm_wmma(const _Float16* __restrict__ A,
                              const _Float16* __restrict__ Bp,
                              const float* __restrict__ dis,
                              float* __restrict__ Hg,
                              float* __restrict__ Ha,
                              int N, int totalTasks) {
  constexpr int K = KT * 32;
  int wave = threadIdx.x >> 5;
  int lane = threadIdx.x & 31;
  int task = blockIdx.x * 8 + wave;
  if (task >= totalTasks) return;          // wave-uniform: EXEC stays all-1s

  int ngroups = N >> 6;                    // 64 cols per wave
  int mtile   = task / ngroups;
  int nbase   = (task % ngroups) * 4;      // 4 n-tiles of 16
  int hi      = lane >> 4;                 // half-wave select
  int mrow    = mtile * 16 + (lane & 15);
  const _Float16* arow = A + (size_t)mrow * K + hi * 16;

  v8f acc[4] = {};
  // prologue loads (kt = 0)
  v8h a0 = *(const v8h*)(arow);
  v8h a1 = *(const v8h*)(arow + 8);
  v8h b0[4], b1[4];
#pragma unroll
  for (int t = 0; t < 4; ++t) {
    const _Float16* p = Bp + (((size_t)(nbase + t) * KT) * 32 + lane) * 16;
    b0[t] = *(const v8h*)(p);
    b1[t] = *(const v8h*)(p + 8);
  }

#pragma unroll
  for (int kt = 0; kt < KT; ++kt) {
    v8h na0, na1, nb0[4], nb1[4];
    if (kt + 1 < KT) {                     // prefetch next K-step
      na0 = *(const v8h*)(arow + (kt + 1) * 32);
      na1 = *(const v8h*)(arow + (kt + 1) * 32 + 8);
#pragma unroll
      for (int t = 0; t < 4; ++t) {
        const _Float16* p =
            Bp + (((size_t)(nbase + t) * KT + (kt + 1)) * 32 + lane) * 16;
        nb0[t] = *(const v8h*)(p);
        nb1[t] = *(const v8h*)(p + 8);
      }
    }
    v16h aa = __builtin_shufflevector(a0, a1, 0, 1, 2, 3, 4, 5, 6, 7,
                                              8, 9, 10, 11, 12, 13, 14, 15);
#pragma unroll
    for (int t = 0; t < 4; ++t) {
      v16h bb = __builtin_shufflevector(b0[t], b1[t], 0, 1, 2, 3, 4, 5, 6, 7,
                                                      8, 9, 10, 11, 12, 13, 14, 15);
      acc[t] = __builtin_amdgcn_wmma_f32_16x16x32_f16(
          /*neg_a=*/false, aa, /*neg_b=*/false, bb,
          /*c_mod=*/(short)0, acc[t], /*reuse_a=*/false, /*reuse_b=*/false);
    }
    if (kt + 1 < KT) {
      a0 = na0; a1 = na1;
#pragma unroll
      for (int t = 0; t < 4; ++t) { b0[t] = nb0[t]; b1[t] = nb1[t]; }
    }
  }

  // C/D layout: VGPR r, lanes 0-15 -> M=r, N=lane; lanes 16-31 -> M=r+8.
  // Store Hg and the fused self-loop term Ha = Hg * dis[row]^2.
  int col   = lane & 15;
  int rbase = mtile * 16 + hi * 8;
#pragma unroll
  for (int r = 0; r < 8; ++r) {
    int row = rbase + r;
    float w = dis[row];
    w *= w;
#pragma unroll
    for (int t = 0; t < 4; ++t) {
      int nc = (nbase + t) * 16 + col;
      float v = acc[t][r];
      Hg[(size_t)row * N + nc] = v;
      Ha[(size_t)row * N + nc] = v * w;
    }
  }
}

// -------------------------------------------------------------- aggregation

// Ha[dst] += Hg[src] * dis[src]*dis[dst]   (N/4 threads per edge, float4)
__global__ void gcn_agg_edges(const long long* __restrict__ src,
                              const long long* __restrict__ dst, long long E,
                              const float* __restrict__ dis,
                              const float* __restrict__ Hg,
                              float* __restrict__ Ha, int N) {
  long long gid = (long long)blockIdx.x * blockDim.x + threadIdx.x;
  int lanesPerEdge = N >> 2;
  long long e = gid / lanesPerEdge;
  if (e >= E) return;
  int c4 = (int)(gid % lanesPerEdge) * 4;
  if (c4 == 0 && e + 2048 < E) {           // stream-prefetch edge list ahead
    __builtin_prefetch(src + e + 2048, 0, 0);
    __builtin_prefetch(dst + e + 2048, 0, 0);
  }
  int s = (int)src[e];
  int d = (int)dst[e];
  float w = dis[s] * dis[d];
  float4 h = *(const float4*)(Hg + (size_t)s * N + c4);
  float* o = Ha + (size_t)d * N + c4;
  atomicAdd(o + 0, h.x * w);
  atomicAdd(o + 1, h.y * w);
  atomicAdd(o + 2, h.z * w);
  atomicAdd(o + 3, h.w * w);
}

// out = (relu?)(Ha + bias); optional f32 output and/or f16 copy for next layer
__global__ void gcn_finalize(const float* __restrict__ Ha,
                             const float* __restrict__ bias,
                             float* __restrict__ outF,
                             _Float16* __restrict__ outH,
                             int N, long long n, int doRelu) {
  long long i = (long long)blockIdx.x * blockDim.x + threadIdx.x;
  if (i >= n) return;
  float v = Ha[i] + bias[(int)(i % N)];
  if (doRelu) v = v > 0.0f ? v : 0.0f;
  if (outF) outF[i] = v;
  if (outH) outH[i] = (_Float16)v;
}

// ------------------------------------------------------------------ driver

extern "C" void kernel_launch(void* const* d_in, const int* in_sizes, int n_in,
                              void* d_out, int out_size, void* d_ws, size_t ws_size,
                              hipStream_t stream) {
  const float*     x  = (const float*)d_in[0];
  const long long* ei = (const long long*)d_in[1];
  const float* W[3] = {(const float*)d_in[2], (const float*)d_in[4],
                       (const float*)d_in[6]};
  const float* bv[3] = {(const float*)d_in[3], (const float*)d_in[5],
                        (const float*)d_in[7]};
  const int Kd[3] = {128, 256, 256};
  const int Nd[3] = {256, 256, 128};

  long long E = (long long)in_sizes[1] / 2;
  const long long* srcI = ei;
  const long long* dstI = ei + E;

  // workspace carve-out
  char* ws = (char*)d_ws;
  size_t off = 0;
  auto carve = [&](size_t bytes) {
    void* p = ws + off;
    off += (bytes + 255) & ~(size_t)255;
    return p;
  };
  float*     dis = (float*)carve((size_t)N_NODES * 4);
  _Float16*  A16 = (_Float16*)carve((size_t)N_NODES * 256 * 2);
  _Float16*  Bp  = (_Float16*)carve((size_t)256 * 256 * 2);
  float*     Hg  = (float*)carve((size_t)N_NODES * 256 * 4);
  float*     Ha  = (float*)carve((size_t)N_NODES * 256 * 4);

  const int B = 256;
  // degrees -> dis
  gcn_zero_f32<<<(N_NODES + B - 1) / B, B, 0, stream>>>(dis, N_NODES);
  gcn_deg_count<<<(int)((E + B - 1) / B), B, 0, stream>>>(dstI, E, dis);
  gcn_deg_rsqrt<<<(N_NODES + B - 1) / B, B, 0, stream>>>(dis, N_NODES);

  // layer-1 input to f16
  {
    long long n4 = (long long)N_NODES * Kd[0] / 4;
    gcn_cvt_f16<<<(int)((n4 + B - 1) / B), B, 0, stream>>>(x, A16, n4);
  }

  for (int l = 0; l < 3; ++l) {
    int K = Kd[l], N = Nd[l];
    // repack weights into B-fragment layout
    gcn_repack_w<<<(K * N + B - 1) / B, B, 0, stream>>>(W[l], Bp, K, N);
    // GEMM (WMMA f16->f32): Hg = A16*W, Ha = Hg*dis^2 (self loop fused)
    int totalTasks = (N_NODES / 16) * (N / 64);
    int gemmBlocks = (totalTasks + 7) / 8;
    if (K == 128) {
      gcn_gemm_wmma<4><<<gemmBlocks, 256, 0, stream>>>(A16, Bp, dis, Hg, Ha,
                                                       N, totalTasks);
    } else {
      gcn_gemm_wmma<8><<<gemmBlocks, 256, 0, stream>>>(A16, Bp, dis, Hg, Ha,
                                                       N, totalTasks);
    }
    // edge aggregation (gather-scale-scatter with f32 atomics)
    long long nElem = (long long)N_NODES * N;
    long long tot = E * (N / 4);
    gcn_agg_edges<<<(int)((tot + B - 1) / B), B, 0, stream>>>(srcI, dstI, E,
                                                              dis, Hg, Ha, N);
    // bias (+relu); layers 0/1 feed next GEMM as f16, layer 2 writes d_out
    if (l < 2) {
      gcn_finalize<<<(int)((nElem + B - 1) / B), B, 0, stream>>>(
          Ha, bv[l], (float*)nullptr, A16, N, nElem, 1);
    } else {
      gcn_finalize<<<(int)((nElem + B - 1) / B), B, 0, stream>>>(
          Ha, bv[l], (float*)d_out, (_Float16*)nullptr, N, nElem, 0);
    }
  }
}
